// DiBiMa_nn_32744830665135
// MI455X (gfx1250) — matrix operand order repaired and checked by reading, not compiled, not run on hardware
//
#include <hip/hip_runtime.h>
#include <hip/hip_bf16.h>

// ---------------------------------------------------------------------------
// DiBiMa (bidirectional Mamba) pipeline for MI455X / gfx1250.
//  * All big GEMMs (in-proj, xW, outW, convd-as-GEMM, sp-conv-as-GEMM) run on
//    v_wmma_f32_16x16x32_f16 (f16 A/B, f32 accum). Weights are staged in LDS
//    per block (8x reduction of global B traffic); A comes straight from
//    global as two b128 loads per lane per k-step.
//  * The selective scan is the sequential bottleneck (2560 steps); it is a
//    register-resident f32 VALU kernel with LDS staging of dt/B/C rows and the
//    dt rank-16 projection fused in-register (dtW row lives in VGPRs).
//  * Backward direction: no reversed activation copies -- in-proj GEMM reads
//    A rows reversed, outW epilogue writes rows reversed and accumulates into
//    the f+b sum buffer.
// ---------------------------------------------------------------------------

typedef __attribute__((ext_vector_type(16))) _Float16 v16h;
typedef __attribute__((ext_vector_type(8)))  _Float16 v8h;
typedef __attribute__((ext_vector_type(8)))  float    v8f;

union AFrag { v16h v; v8h h[2]; };

__device__ __forceinline__ float siluf(float x) { return x / (1.f + __expf(-x)); }
__device__ __forceinline__ float softplusf(float x) {
  return (x > 20.f) ? x : __logf(1.f + __expf(x));
}

// ============================ WMMA GEMM ====================================
// D[m,n] = sum_j sum_k A[(arow(m)+j), k] * Bw_j[n, k]
// A f16 row-major [*, lda]; Bw f16 row-major [N, ldb] (j taps jstrideB apart).
// Block: 8 waves => 128 M x 64 N. B tile staged in LDS, shared by all waves.
struct GemmP {
  const _Float16* A;
  const _Float16* Bw;
  float*    out_f32;
  _Float16* out_f16;
  const float* xn;      // epi2 residual
  float*       mo;      // epi2 accumulation target
  const float* bias;    // epi3/4
  const float* bn_g; const float* bn_b; const float* bn_m; const float* bn_v; // epi3
  int M, N, K, lda, ldb, ldd;
  int jcount, jstrideB; // conv taps: A advances 1 row per tap, B advances jstrideB
  int T, Tpad;          // Tpad!=0 -> A rows live in padded [B, Tpad, lda] layout
  int arev, orev;       // reverse A rows / output rows within each sequence
};

#define SB_STRIDE 40   // 64 rows x 32 halfs, padded to 40 (80B) to spread LDS banks

template <int NT, int EPI>
__global__ __launch_bounds__(256) void wmma_gemm_kernel(GemmP p) {
  __shared__ _Float16 sB[64 * SB_STRIDE];

  const int lane = threadIdx.x & 31;
  const int wave = threadIdx.x >> 5;
  const int lm   = lane & 15;
  const int g    = lane >> 4;

  const int tileM0 = blockIdx.x * 128 + wave * 16;   // 16-row tile, single sequence
  const int n0base = blockIdx.y * 64;
  const int b      = tileM0 / p.T;
  const int tbase  = tileM0 % p.T;

  long arowBase;
  if (p.Tpad)      arowBase = (long)b * p.Tpad + tbase;          // taps t..t+2 via +j
  else if (p.arev) arowBase = (long)b * p.T + (p.T - 1 - tbase); // lane row = base - lm
  else             arowBase = tileM0;
  const long arow = p.arev ? (arowBase - lm) : (arowBase + lm);

  // cooperative B staging: thread -> (row 0..63, 8-half chunk 0..3)
  const int srow = threadIdx.x >> 2;
  const int skc  = (threadIdx.x & 3) * 8;
  const _Float16* bstage = p.Bw + (long)(n0base + srow) * p.ldb + skc;
  _Float16* sdst = &sB[srow * SB_STRIDE + skc];

  v8f acc[NT] = {};

  for (int j = 0; j < p.jcount; ++j) {
    const _Float16* ap = p.A + (arow + j) * (long)p.lda + 8 * g;
    const _Float16* bs = bstage + (long)j * p.jstrideB;
    for (int k0 = 0; k0 < p.K; k0 += 32) {
      __syncthreads();                                  // previous reads done
      *(v8h*)sdst = *(const v8h*)(bs + k0);             // global -> LDS (b128)
      AFrag fa;
      fa.h[0] = *(const v8h*)(ap);                      // K = k0 + 8g      .. +7
      fa.h[1] = *(const v8h*)(ap + 16);                 // K = k0 + 16 + 8g .. +7
      __builtin_prefetch(ap + 256, 0, 3);
      ap += 32;
      __syncthreads();                                  // LDS tile visible
#pragma unroll
      for (int nt = 0; nt < NT; ++nt) {
        const _Float16* sp2 = &sB[(nt * 16 + lm) * SB_STRIDE + 8 * g];
        AFrag fb;
        fb.h[0] = *(const v8h*)sp2;                     // ds_load_b128
        fb.h[1] = *(const v8h*)(sp2 + 16);
        acc[nt] = __builtin_amdgcn_wmma_f32_16x16x32_f16(
            false, fa.v, false, fb.v, (short)0, acc[nt], false, false);
      }
    }
  }

#pragma unroll
  for (int nt = 0; nt < NT; ++nt) {
    const int n = n0base + nt * 16 + lm;
#pragma unroll
    for (int i = 0; i < 8; ++i) {
      const int  r8   = i + 8 * g;        // row within 16-row tile
      const long mrow = tileM0 + r8;
      const int  t    = tbase + r8;
      float v = acc[nt][i];
      if constexpr (EPI == 0) {                  // f16 store (in-proj -> xz)
        p.out_f16[mrow * p.ldd + n] = (_Float16)v;
      } else if constexpr (EPI == 1) {           // f32 store (xW -> dbc)
        p.out_f32[mrow * p.ldd + n] = v;
      } else if constexpr (EPI == 2) {           // out-proj + residual (+ f/b sum)
        const long rr = p.orev ? ((long)b * p.T + (p.T - 1 - t)) : mrow;
        const float r = v + p.xn[rr * 256 + n];
        if (p.orev) p.mo[rr * 256 + n] += r; else p.mo[rr * 256 + n] = r;
      } else if constexpr (EPI == 3) {           // convd: bias + BN + silu -> f16 pad
        float xv = v + p.bias[n];
        xv = (xv - p.bn_m[n]) * rsqrtf(p.bn_v[n] + 1e-5f) * p.bn_g[n] + p.bn_b[n];
        xv = siluf(xv);
        p.out_f16[((long)b * p.Tpad + t + 1) * (long)p.ldd + n] = (_Float16)xv;
      } else {                                   // sp conv: bias + pixel shuffle (R=2)
        const float xv = v + p.bias[n];
        p.out_f32[((long)(b * 64 + (n >> 1))) * (2L * p.T) + 2L * t + (n & 1)] = xv;
      }
    }
  }
}

// ======================= fp32 encoder conv (k=3, same) =====================
__global__ __launch_bounds__(256) void conv1d_silu_f32(
    const float* __restrict__ x, const float* __restrict__ w,
    const float* __restrict__ bias, float* __restrict__ y, int Cin, int T) {
  const int t  = blockIdx.x * 256 + threadIdx.x;
  const int co = blockIdx.y;
  const int b  = blockIdx.z;
  const int Cout = gridDim.y;
  if (t >= T) return;
  float acc = bias[co];
  const float* wr = w + (long)co * Cin * 3;
  const float* xb = x + (long)b * Cin * T;
  for (int ci = 0; ci < Cin; ++ci) {
    const float* xr = xb + (long)ci * T + t;
    const float w0 = wr[ci * 3 + 0], w1 = wr[ci * 3 + 1], w2 = wr[ci * 3 + 2];
    if (t > 0)     acc += w0 * xr[-1];
    acc += w1 * xr[0];
    if (t < T - 1) acc += w2 * xr[1];
  }
  y[((long)b * Cout + co) * T + t] = siluf(acc);
}

// ============ RMSNorm #1 + NCH->NHC transpose + f16 copy (wave/row) ========
__global__ __launch_bounds__(256) void rms1_kernel(
    const float* __restrict__ e, const float* __restrict__ w1,
    float* __restrict__ t_act, float* __restrict__ xn,
    _Float16* __restrict__ xn_h, int T, int BT) {
  const int wid  = (int)((blockIdx.x * 256 + threadIdx.x) >> 5);
  const int lane = threadIdx.x & 31;
  if (wid >= BT) return;
  const int b = wid / T, t = wid % T;
  const float* ep = e + (long)b * 256 * T + t;
  float v[8], ss = 0.f;
#pragma unroll
  for (int k = 0; k < 8; ++k) {
    v[k] = ep[(long)(lane + 32 * k) * T];
    ss += v[k] * v[k];
  }
  for (int off = 16; off; off >>= 1) ss += __shfl_xor(ss, off, 32);
  const float scale = rsqrtf(ss * (1.f / 256.f) + 1e-6f);
  const long r = (long)wid * 256;
#pragma unroll
  for (int k = 0; k < 8; ++k) {
    const int c = lane + 32 * k;
    t_act[r + c] = v[k];
    const float xv = v[k] * scale * w1[c];
    xn[r + c] = xv;
    xn_h[r + c] = (_Float16)xv;
  }
}

// ============ RMSNorm #2 + concat [combined | t_act] -> padded f16 =========
__global__ __launch_bounds__(256) void rms2_kernel(
    const float* __restrict__ mo, const float* __restrict__ t_act,
    const float* __restrict__ w2, _Float16* __restrict__ m_pad, int T, int BT) {
  const int wid  = (int)((blockIdx.x * 256 + threadIdx.x) >> 5);
  const int lane = threadIdx.x & 31;
  if (wid >= BT) return;
  const int b = wid / T, t = wid % T;
  const long r = (long)wid * 256;
  float v[8], ss = 0.f;
#pragma unroll
  for (int k = 0; k < 8; ++k) { v[k] = mo[r + lane + 32 * k]; ss += v[k] * v[k]; }
  for (int off = 16; off; off >>= 1) ss += __shfl_xor(ss, off, 32);
  const float scale = rsqrtf(ss * (1.f / 256.f) + 1e-6f);
  const long orow = ((long)b * (T + 2) + t + 1) * 512;
#pragma unroll
  for (int k = 0; k < 8; ++k) {
    const int c = lane + 32 * k;
    m_pad[orow + c]       = (_Float16)(v[k] * scale * w2[c]);
    m_pad[orow + 256 + c] = (_Float16)t_act[r + c];
  }
}

// =================== depthwise causal conv (k=4) + silu ====================
__global__ __launch_bounds__(256) void dwconv_silu_kernel(
    const _Float16* __restrict__ xz_h, const float* __restrict__ w,
    const float* __restrict__ cb, _Float16* __restrict__ xcs, int T) {
  const long gid = (long)blockIdx.x * 256 + threadIdx.x; // over M*512
  const int  d = (int)(gid & 511);
  const long m = gid >> 9;
  const int  t = (int)(m % T);
  float acc = cb[d];
#pragma unroll
  for (int j = 0; j < 4; ++j) {
    const int tt = t - 3 + j;
    if (tt >= 0) acc += w[d * 4 + j] * (float)xz_h[(m - 3 + j) * 1024 + d];
  }
  xcs[m * 512 + d] = (_Float16)siluf(acc);
}

// ====================== selective scan (sequential in T) ===================
// thread = (b, d); h[16] lives in registers across all T steps; dt projection
// (rank 16) fused with dtW row held in registers; dt/B/C rows staged in LDS.
#define SCAN_TC 64
__global__ __launch_bounds__(256) void scan_kernel(
    const _Float16* __restrict__ xcs, const _Float16* __restrict__ xz_h,
    const float* __restrict__ dbc, const float* __restrict__ dtW,
    const float* __restrict__ dtb, const float* __restrict__ Alog,
    const float* __restrict__ Dp, _Float16* __restrict__ ymul, int T) {
  __shared__ float sbc[SCAN_TC * 48];
  const int b = blockIdx.x >> 1;
  const int d = ((blockIdx.x & 1) << 8) + threadIdx.x;

  float A[16], h[16], wdt[16];
#pragma unroll
  for (int n = 0; n < 16; ++n) { A[n] = -__expf(Alog[d * 16 + n]); h[n] = 0.f; }
#pragma unroll
  for (int i = 0; i < 16; ++i) wdt[i] = dtW[d * 16 + i];
  const float bdt = dtb[d], Dv = Dp[d];
  const long rowbase = (long)b * T;

  for (int t0 = 0; t0 < T; t0 += SCAN_TC) {
    __syncthreads();
    const float* src = dbc + (rowbase + t0) * 48;
    for (int idx = threadIdx.x; idx < SCAN_TC * 48; idx += 256) sbc[idx] = src[idx];
    __syncthreads();
    for (int tl = 0; tl < SCAN_TC; ++tl) {
      const long row = rowbase + t0 + tl;
      const float u  = (float)xcs[row * 512 + d];
      const float zv = (float)xz_h[row * 1024 + 512 + d];
      const float* S = sbc + tl * 48;          // [0:16)=dt-rank, [16:32)=B, [32:48)=C
      float dtv = bdt;
#pragma unroll
      for (int i = 0; i < 16; ++i) dtv += S[i] * wdt[i];
      dtv = softplusf(dtv);
      const float du = dtv * u;
      float acc = 0.f;
#pragma unroll
      for (int n = 0; n < 16; ++n) {
        const float dA = __expf(dtv * A[n]);
        h[n] = dA * h[n] + du * S[16 + n];
        acc += h[n] * S[32 + n];
      }
      const float y = acc + Dv * u;
      ymul[row * 512 + d] = (_Float16)(y * siluf(zv));
    }
  }
}

// ============================ weight converters ============================
__global__ __launch_bounds__(256) void cvt_f16(const float* __restrict__ s,
                                               _Float16* __restrict__ d, long n) {
  for (long i = (long)blockIdx.x * 256 + threadIdx.x; i < n; i += (long)gridDim.x * 256)
    d[i] = (_Float16)s[i];
}

// (N,K,3) fp32 -> [3][N][K] f16 (per-tap row-major weights for conv-as-GEMM)
__global__ __launch_bounds__(256) void convw_cvt(const float* __restrict__ w,
                                                 _Float16* __restrict__ dst,
                                                 int N, int K) {
  const long total = (long)N * K * 3;
  const long i = (long)blockIdx.x * 256 + threadIdx.x;
  if (i >= total) return;
  const int  j  = (int)(i % 3);
  const long nk = i / 3;
  const int  k  = (int)(nk % K);
  const int  n  = (int)(nk / K);
  dst[((long)j * N + n) * K + k] = (_Float16)w[i];
}

// zero the halo rows of the padded conv input buffers
__global__ __launch_bounds__(512) void zero_pads(_Float16* m_pad, _Float16* d_pad, int T) {
  const int b = blockIdx.x, i = threadIdx.x;
  const long r0 = (long)b * (T + 2), r1 = r0 + T + 1;
  m_pad[r0 * 512 + i] = (_Float16)0.f;
  m_pad[r1 * 512 + i] = (_Float16)0.f;
  if (i < 256) {
    d_pad[r0 * 256 + i] = (_Float16)0.f;
    d_pad[r1 * 256 + i] = (_Float16)0.f;
  }
}

// ================================ driver ===================================
extern "C" void kernel_launch(void* const* d_in, const int* in_sizes, int n_in,
                              void* d_out, int out_size, void* d_ws, size_t ws_size,
                              hipStream_t stream) {
  (void)in_sizes; (void)n_in; (void)out_size; (void)ws_size;
  constexpr int Bsz = 16, T = 2560, Tp = T + 2;
  constexpr long M = (long)Bsz * T; // 40960

  const float* x       = (const float*)d_in[0];
  const float* enc_w1  = (const float*)d_in[1];
  const float* enc_b1  = (const float*)d_in[2];
  const float* enc_w2  = (const float*)d_in[3];
  const float* enc_b2  = (const float*)d_in[4];
  const float* norm1_w = (const float*)d_in[5];
  const float* norm2_w = (const float*)d_in[6];
  const float* dir_inW[2]   = { (const float*)d_in[7],  (const float*)d_in[16] };
  const float* dir_convW[2] = { (const float*)d_in[8],  (const float*)d_in[17] };
  const float* dir_convb[2] = { (const float*)d_in[9],  (const float*)d_in[18] };
  const float* dir_xW[2]    = { (const float*)d_in[10], (const float*)d_in[19] };
  const float* dir_dtW[2]   = { (const float*)d_in[11], (const float*)d_in[20] };
  const float* dir_dtb[2]   = { (const float*)d_in[12], (const float*)d_in[21] };
  const float* dir_Alog[2]  = { (const float*)d_in[13], (const float*)d_in[22] };
  const float* dir_D[2]     = { (const float*)d_in[14], (const float*)d_in[23] };
  const float* dir_outW[2]  = { (const float*)d_in[15], (const float*)d_in[24] };
  const float* convd_w = (const float*)d_in[25];
  const float* convd_b = (const float*)d_in[26];
  const float* bn_g    = (const float*)d_in[27];
  const float* bn_b    = (const float*)d_in[28];
  const float* bn_mean = (const float*)d_in[29];
  const float* bn_var  = (const float*)d_in[30];
  const float* sp_w    = (const float*)d_in[31];
  const float* sp_b    = (const float*)d_in[32];

  char* ws = (char*)d_ws;
  size_t off = 0;
  auto alloc = [&](size_t bytes) -> void* {
    void* p = ws + off;
    off = (off + bytes + 255) & ~(size_t)255;
    return p;
  };

  float*     e1      = (float*)    alloc((size_t)Bsz * 128 * T * 4);
  float*     e2      = (float*)    alloc((size_t)Bsz * 256 * T * 4);
  float*     t_act   = (float*)    alloc((size_t)M * 256 * 4);
  float*     xn      = (float*)    alloc((size_t)M * 256 * 4);
  _Float16*  xn_h    = (_Float16*) alloc((size_t)M * 256 * 2);
  float*     mo_sum  = (float*)    alloc((size_t)M * 256 * 4);
  _Float16*  xz_h    = (_Float16*) alloc((size_t)M * 1024 * 2);
  _Float16*  xcs     = (_Float16*) alloc((size_t)M * 512 * 2);
  float*     dbc     = (float*)    alloc((size_t)M * 48 * 4);
  _Float16*  ymul    = (_Float16*) alloc((size_t)M * 512 * 2);
  _Float16*  m_pad   = (_Float16*) alloc((size_t)Bsz * Tp * 512 * 2);
  _Float16*  d_padb  = (_Float16*) alloc((size_t)Bsz * Tp * 256 * 2);
  _Float16*  wIn_h   = (_Float16*) alloc((size_t)1024 * 256 * 2);
  _Float16*  wX_h    = (_Float16*) alloc((size_t)64 * 512 * 2);   // padded 48->64 rows
  _Float16*  wO_h    = (_Float16*) alloc((size_t)256 * 512 * 2);
  _Float16*  wCd_h   = (_Float16*) alloc((size_t)3 * 256 * 512 * 2);
  _Float16*  wSp_h   = (_Float16*) alloc((size_t)3 * 128 * 256 * 2);

  // ---- encoder ----
  conv1d_silu_f32<<<dim3((T + 255) / 256, 128, Bsz), 256, 0, stream>>>(x, enc_w1, enc_b1, e1, 64, T);
  conv1d_silu_f32<<<dim3((T + 255) / 256, 256, Bsz), 256, 0, stream>>>(e1, enc_w2, enc_b2, e2, 128, T);
  rms1_kernel<<<(int)(M / 8), 256, 0, stream>>>(e2, norm1_w, t_act, xn, xn_h, T, (int)M);

  // ---- bidirectional mamba (forward = 0, backward = 1; scratch reused) ----
  for (int dir = 0; dir < 2; ++dir) {
    cvt_f16<<<1024, 256, 0, stream>>>(dir_inW[dir],  wIn_h, 1024L * 256);
    cvt_f16<<<96,   256, 0, stream>>>(dir_xW[dir],   wX_h,  48L * 512);
    cvt_f16<<<512,  256, 0, stream>>>(dir_outW[dir], wO_h,  256L * 512);

    GemmP p = {};
    // in-projection: xz = xn(rev?) @ inW^T   -> f16 [M,1024]
    p.A = xn_h; p.Bw = wIn_h; p.out_f16 = xz_h;
    p.M = (int)M; p.N = 1024; p.K = 256; p.lda = 256; p.ldb = 256; p.ldd = 1024;
    p.jcount = 1; p.jstrideB = 0; p.T = T; p.Tpad = 0; p.arev = dir; p.orev = 0;
    wmma_gemm_kernel<4, 0><<<dim3((int)(M / 128), 16), 256, 0, stream>>>(p);

    // depthwise causal conv + silu
    dwconv_silu_kernel<<<(int)(M * 512 / 256), 256, 0, stream>>>(
        xz_h, dir_convW[dir], dir_convb[dir], xcs, T);

    // dbc = xc_silu @ xW^T  -> f32 [M,48]  (NT=3: 48 = 3 x 16 valid col-tiles)
    GemmP q = {};
    q.A = xcs; q.Bw = wX_h; q.out_f32 = dbc;
    q.M = (int)M; q.N = 48; q.K = 512; q.lda = 512; q.ldb = 512; q.ldd = 48;
    q.jcount = 1; q.T = T;
    wmma_gemm_kernel<3, 1><<<dim3((int)(M / 128), 1), 256, 0, stream>>>(q);

    // selective scan (fuses dt projection + softplus + gate)
    scan_kernel<<<Bsz * 2, 256, 0, stream>>>(
        xcs, xz_h, dbc, dir_dtW[dir], dir_dtb[dir], dir_Alog[dir], dir_D[dir], ymul, T);

    // out-projection + residual, accumulate f+b (backward writes reversed rows)
    GemmP r = {};
    r.A = ymul; r.Bw = wO_h; r.xn = xn; r.mo = mo_sum;
    r.M = (int)M; r.N = 256; r.K = 512; r.lda = 512; r.ldb = 512; r.ldd = 256;
    r.jcount = 1; r.T = T; r.arev = 0; r.orev = dir;
    wmma_gemm_kernel<4, 2><<<dim3((int)(M / 128), 4), 256, 0, stream>>>(r);
  }

  // ---- combine: rmsnorm(mo_f + mo_b) || t_act  -> padded f16 [B,Tp,512] ----
  rms2_kernel<<<(int)(M / 8), 256, 0, stream>>>(mo_sum, t_act, norm2_w, m_pad, T, (int)M);
  zero_pads<<<Bsz, 512, 0, stream>>>(m_pad, d_padb, T);

  // ---- convd (512->256, k=3) as 3-tap WMMA GEMM + BN + silu ----
  convw_cvt<<<(int)((3L * 256 * 512 + 255) / 256), 256, 0, stream>>>(convd_w, wCd_h, 256, 512);
  {
    GemmP c = {};
    c.A = m_pad; c.Bw = wCd_h; c.out_f16 = d_padb;
    c.bias = convd_b; c.bn_g = bn_g; c.bn_b = bn_b; c.bn_m = bn_mean; c.bn_v = bn_var;
    c.M = (int)M; c.N = 256; c.K = 512; c.lda = 512; c.ldb = 512; c.ldd = 256;
    c.jcount = 3; c.jstrideB = 256 * 512; c.T = T; c.Tpad = Tp;
    wmma_gemm_kernel<4, 3><<<dim3((int)(M / 128), 4), 256, 0, stream>>>(c);
  }

  // ---- sp conv (256->128, k=3) as 3-tap WMMA GEMM + pixel-shuffle store ----
  convw_cvt<<<(int)((3L * 128 * 256 + 255) / 256), 256, 0, stream>>>(sp_w, wSp_h, 128, 256);
  {
    GemmP s = {};
    s.A = d_padb; s.Bw = wSp_h; s.out_f32 = (float*)d_out; s.bias = sp_b;
    s.M = (int)M; s.N = 128; s.K = 256; s.lda = 256; s.ldb = 256; s.ldd = 0;
    s.jcount = 3; s.jstrideB = 128 * 256; s.T = T; s.Tpad = Tp;
    wmma_gemm_kernel<4, 4><<<dim3((int)(M / 128), 2), 256, 0, stream>>>(s);
  }
}
